// GroupedExperts_54219667145005
// MI455X (gfx1250) — matrix operand order repaired and checked by reading, not compiled, not run on hardware
//
#include <hip/hip_runtime.h>
#include <hip/hip_bf16.h>

#define N_TOKENS  16384
#define DIM       2048
#define HIDDEN    1408
#define N_EXPERTS 8
#define TOK_PER_E (N_TOKENS / N_EXPERTS)   // 2048

typedef __bf16 bf16;
typedef __attribute__((ext_vector_type(4)))  int    v4i;
typedef __attribute__((ext_vector_type(4)))  float  v4f;
typedef __attribute__((ext_vector_type(8)))  float  v8f;
typedef __attribute__((ext_vector_type(4)))  __bf16 v4bf;
typedef __attribute__((ext_vector_type(8)))  __bf16 v8bf;
typedef __attribute__((ext_vector_type(16))) __bf16 v16bf;

typedef __attribute__((address_space(1))) v4i* gptr_b128;  // global int4*
typedef __attribute__((address_space(3))) v4i* lptr_b128;  // LDS int4*

#define BM  128
#define BN  128   // stage-1 N tile
#define BN2 256   // stage-2 N tile
#define BK  32
#define LDSS 40   // BK + 8 elements: 80B row pitch, 16B aligned, conflict-free frag reads

#if __has_builtin(__builtin_amdgcn_global_load_async_to_lds_b128)
#define HAVE_ASYNC_LDS 1
#else
#define HAVE_ASYNC_LDS 0
#endif

__device__ __forceinline__ void wait_async_lds() {
#if HAVE_ASYNC_LDS
#if __has_builtin(__builtin_amdgcn_s_wait_asynccnt)
  __builtin_amdgcn_s_wait_asynccnt(0);
#else
  asm volatile("s_wait_asynccnt 0x0" ::: "memory");
#endif
#endif
}

// A-fragment (16x32 bf16): lane<16 holds row r, K {k0..k0+7} in v0..3 and
// K {16+k0..16+k0+7} in v4..7 with k0 = (lane>>4)*8.
__device__ __forceinline__ v16bf load_fragA(const bf16* rowp, int lane) {
  const int k0 = (lane >> 4) * 8;
  v8bf lo = *(const v8bf*)(rowp + k0);
  v8bf hi = *(const v8bf*)(rowp + 16 + k0);
  v16bf r;
#pragma unroll
  for (int i = 0; i < 8; ++i) { r[i] = lo[i]; r[8 + i] = hi[i]; }
  return r;
}

// B-fragment (32x16 bf16), loaded from B^T rows (n-major, K contiguous):
// lane<16 holds column n, K {k0..k0+7} in v0..3 and K {16+k0..16+k0+7} in v4..7
// with k0 = (lane>>4)*8.
__device__ __forceinline__ v16bf load_fragB(const bf16* rowp, int lane) {
  const int k0 = (lane >> 4) * 8;
  v8bf lo = *(const v8bf*)(rowp + k0);
  v8bf hi = *(const v8bf*)(rowp + 16 + k0);
  v16bf r;
#pragma unroll
  for (int i = 0; i < 8; ++i) { r[i] = lo[i]; r[8 + i] = hi[i]; }
  return r;
}

__device__ __forceinline__ void cvt_store4(bf16* dst, v4f v) {
  v4bf b;
#pragma unroll
  for (int i = 0; i < 4; ++i) b[i] = (bf16)v[i];
  *(v4bf*)dst = b;
}

// ---------------- Stage 1: h = silu(x @ W1^T) * (x @ W3^T), bf16 out ----------------
__global__ __launch_bounds__(256)
void moe_stage1(const float* __restrict__ x, const float* __restrict__ w1,
                const float* __restrict__ w3, bf16* __restrict__ hb) {
  __shared__ bf16 sA[2][BM * LDSS];
  __shared__ bf16 sG[2][BN * LDSS];
  __shared__ bf16 sU[2][BN * LDSS];

  const int e  = blockIdx.z;
  const int m0 = blockIdx.y * BM;
  const int n0 = blockIdx.x * BN;

  const float* xe  = x  + (size_t)(e * TOK_PER_E + m0) * DIM;
  const float* w1e = w1 + (size_t)e * HIDDEN * DIM + (size_t)n0 * DIM;
  const float* w3e = w3 + (size_t)e * HIDDEN * DIM + (size_t)n0 * DIM;

  const int t    = threadIdx.x;
  const int lane = t & 31;
  const int wave = t >> 5;
  const int wm   = (wave & 1) * 64;   // 2 waves along M (64 rows each)
  const int wn   = (wave >> 1) * 32;  // 4 waves along N (32 cols each)
  const int r16  = lane & 15;

  v8f accg[4][2], accu[4][2];
#pragma unroll
  for (int mi = 0; mi < 4; ++mi)
#pragma unroll
    for (int ni = 0; ni < 2; ++ni)
#pragma unroll
      for (int j = 0; j < 8; ++j) { accg[mi][ni][j] = 0.f; accu[mi][ni][j] = 0.f; }

  const int lrow = t >> 3;          // 0..31
  const int lcol = (t & 7) * 4;     // 0,4,...,28

  v4f ra[4], rg[4], ru[4];

  auto load_tiles = [&](int kk) {
#pragma unroll
    for (int i = 0; i < 4; ++i) {
      const size_t roff = (size_t)(lrow + 32 * i);
      ra[i] = *(const v4f*)(xe  + roff * DIM + kk + lcol);
      rg[i] = *(const v4f*)(w1e + roff * DIM + kk + lcol);
      ru[i] = *(const v4f*)(w3e + roff * DIM + kk + lcol);
    }
  };
  auto store_tiles = [&](int b) {
#pragma unroll
    for (int i = 0; i < 4; ++i) {
      const int roff = (lrow + 32 * i) * LDSS + lcol;
      cvt_store4(&sA[b][roff], ra[i]);
      cvt_store4(&sG[b][roff], rg[i]);
      cvt_store4(&sU[b][roff], ru[i]);
    }
  };
  auto compute = [&](int b) {
    v16bf af[4], gf[2], uf[2];
#pragma unroll
    for (int i = 0; i < 4; ++i)
      af[i] = load_fragA(&sA[b][(wm + i * 16 + r16) * LDSS], lane);
#pragma unroll
    for (int i = 0; i < 2; ++i) {
      gf[i] = load_fragB(&sG[b][(wn + i * 16 + r16) * LDSS], lane);
      uf[i] = load_fragB(&sU[b][(wn + i * 16 + r16) * LDSS], lane);
    }
#pragma unroll
    for (int mi = 0; mi < 4; ++mi)
#pragma unroll
      for (int ni = 0; ni < 2; ++ni) {
        accg[mi][ni] = __builtin_amdgcn_wmma_f32_16x16x32_bf16(
            false, af[mi], false, gf[ni], (short)0, accg[mi][ni], false, false);
        accu[mi][ni] = __builtin_amdgcn_wmma_f32_16x16x32_bf16(
            false, af[mi], false, uf[ni], (short)0, accu[mi][ni], false, false);
      }
  };

  // Software pipeline: prefetch k+1 to regs, WMMA on buf cur, stage into buf nxt.
  load_tiles(0);
  store_tiles(0);
  __syncthreads();
  for (int kk = 0; kk < DIM; kk += BK) {
    const int cur = (kk / BK) & 1;
    const bool more = (kk + BK) < DIM;
    if (more) load_tiles(kk + BK);
    compute(cur);
    if (more) store_tiles(cur ^ 1);
    __syncthreads();
  }

  // SwiGLU epilogue. C/D layout: row = 8*(lane>=16)+vgpr, col = lane&15.
  bf16* he = hb + (size_t)(e * TOK_PER_E + m0) * HIDDEN + n0;
  const int mlo = (lane >> 4) * 8;
#pragma unroll
  for (int mi = 0; mi < 4; ++mi)
#pragma unroll
    for (int ni = 0; ni < 2; ++ni) {
      v8f g = accg[mi][ni];
      v8f u = accu[mi][ni];
#pragma unroll
      for (int j = 0; j < 8; ++j) {
        const float gv = g[j];
        const float s  = __builtin_amdgcn_rcpf(1.0f + __expf(-gv));  // v_rcp_f32
        const float hv = gv * s * u[j];
        he[(size_t)(wm + mi * 16 + mlo + j) * HIDDEN + (wn + ni * 16 + r16)] = (bf16)hv;
      }
    }
}

// ---------------- Stage 2: out = h @ W2^T, f32 out ----------------
__global__ __launch_bounds__(256)
void moe_stage2(const bf16* __restrict__ hb, const float* __restrict__ w2,
                float* __restrict__ out) {
  __shared__ bf16 sA[2][BM * LDSS];
  __shared__ bf16 sB[2][BN2 * LDSS];

  const int e  = blockIdx.z;
  const int m0 = blockIdx.y * BM;
  const int n0 = blockIdx.x * BN2;

  const bf16*  he  = hb + (size_t)(e * TOK_PER_E + m0) * HIDDEN;
  const float* w2e = w2 + (size_t)e * DIM * HIDDEN + (size_t)n0 * HIDDEN;

  const int t    = threadIdx.x;
  const int lane = t & 31;
  const int wave = t >> 5;
  const int wm   = (wave & 1) * 64;   // 2 waves along M
  const int wn   = (wave >> 1) * 64;  // 4 waves along N (64 cols each)
  const int r16  = lane & 15;

  v8f acc[4][4];
#pragma unroll
  for (int mi = 0; mi < 4; ++mi)
#pragma unroll
    for (int ni = 0; ni < 4; ++ni)
#pragma unroll
      for (int j = 0; j < 8; ++j) acc[mi][ni][j] = 0.f;

  const int lrow8 = t >> 2;         // 0..63, bf16 A copy: 16B per lane
  const int c8    = (t & 3) * 8;    // 0,8,16,24
  const int lrow  = t >> 3;         // 0..31, fp32 B tile
  const int lcol  = (t & 7) * 4;

#if HAVE_ASYNC_LDS
  // gfx1250 async global->LDS copy for the (already bf16) h tile: no VGPR
  // round-trip, tracked on ASYNCcnt, overlaps the WMMA burst on buf `cur`.
  auto stageA_issue = [&](int b, int kk) {
#pragma unroll
    for (int i = 0; i < 2; ++i) {
      const bf16* g = he + (size_t)(lrow8 + 64 * i) * HIDDEN + kk + c8;
      bf16* l = &sA[b][(lrow8 + 64 * i) * LDSS + c8];
      __builtin_amdgcn_global_load_async_to_lds_b128(
          (gptr_b128)g, (lptr_b128)l, 0, 0);
    }
  };
  auto stageA_commit = [&](int b) { (void)b; };
#else
  v8bf rha[2];
  auto stageA_issue = [&](int b, int kk) {
    (void)b;
#pragma unroll
    for (int i = 0; i < 2; ++i)
      rha[i] = *(const v8bf*)(he + (size_t)(lrow8 + 64 * i) * HIDDEN + kk + c8);
  };
  auto stageA_commit = [&](int b) {
#pragma unroll
    for (int i = 0; i < 2; ++i)
      *(v8bf*)(&sA[b][(lrow8 + 64 * i) * LDSS + c8]) = rha[i];
  };
#endif

  v4f rb[8];
  auto loadB = [&](int kk) {
#pragma unroll
    for (int i = 0; i < 8; ++i)
      rb[i] = *(const v4f*)(w2e + (size_t)(lrow + 32 * i) * HIDDEN + kk + lcol);
  };
  auto storeB = [&](int b) {
#pragma unroll
    for (int i = 0; i < 8; ++i)
      cvt_store4(&sB[b][(lrow + 32 * i) * LDSS + lcol], rb[i]);
  };
  auto compute = [&](int b) {
    v16bf af[4], bfg[4];
#pragma unroll
    for (int i = 0; i < 4; ++i)
      af[i] = load_fragA(&sA[b][(wm + i * 16 + r16) * LDSS], lane);
#pragma unroll
    for (int i = 0; i < 4; ++i)
      bfg[i] = load_fragB(&sB[b][(wn + i * 16 + r16) * LDSS], lane);
#pragma unroll
    for (int mi = 0; mi < 4; ++mi)
#pragma unroll
      for (int ni = 0; ni < 4; ++ni)
        acc[mi][ni] = __builtin_amdgcn_wmma_f32_16x16x32_bf16(
            false, af[mi], false, bfg[ni], (short)0, acc[mi][ni], false, false);
  };

  // Pipeline: async A copy + B reg-stage into buf nxt while WMMA runs on cur.
  stageA_issue(0, 0);
  loadB(0);
  stageA_commit(0);
  storeB(0);
  wait_async_lds();
  __syncthreads();
  for (int kk = 0; kk < HIDDEN; kk += BK) {
    const int cur = (kk / BK) & 1;
    const bool more = (kk + BK) < HIDDEN;
    if (more) {
      stageA_issue(cur ^ 1, kk + BK);
      loadB(kk + BK);
    }
    compute(cur);
    if (more) {
      stageA_commit(cur ^ 1);
      storeB(cur ^ 1);
    }
    wait_async_lds();
    __syncthreads();
  }

  float* oe = out + (size_t)(e * TOK_PER_E + m0) * DIM + n0;
  const int mlo = (lane >> 4) * 8;
#pragma unroll
  for (int mi = 0; mi < 4; ++mi)
#pragma unroll
    for (int ni = 0; ni < 4; ++ni) {
      v8f a = acc[mi][ni];
#pragma unroll
      for (int j = 0; j < 8; ++j)
        oe[(size_t)(wm + mi * 16 + mlo + j) * DIM + (wn + ni * 16 + r16)] = a[j];
    }
}

extern "C" void kernel_launch(void* const* d_in, const int* in_sizes, int n_in,
                              void* d_out, int out_size, void* d_ws, size_t ws_size,
                              hipStream_t stream) {
  // setup_inputs order: x, w1, w2, w3, num_tokens_per_expert
  const float* x  = (const float*)d_in[0];
  const float* w1 = (const float*)d_in[1];
  const float* w2 = (const float*)d_in[2];
  const float* w3 = (const float*)d_in[3];
  float* out = (float*)d_out;
  bf16* hb = (bf16*)d_ws;   // 16384 x 1408 bf16 = ~46.1 MB intermediate

  dim3 blk(256, 1, 1);
  dim3 g1(HIDDEN / BN,  TOK_PER_E / BM, N_EXPERTS);  // 11 x 16 x 8
  dim3 g2(DIM    / BN2, TOK_PER_E / BM, N_EXPERTS);  //  8 x 16 x 8
  moe_stage1<<<g1, blk, 0, stream>>>(x, w1, w3, hb);
  moe_stage2<<<g2, blk, 0, stream>>>(hb, w2, out);
}
